// LongFormer_18391049961867
// MI455X (gfx1250) — compile-verified, matrix operand order checked by reading, not run on hardware
//
#include <hip/hip_runtime.h>
#include <hip/hip_bf16.h>
#include <math.h>

// ---------------------------------------------------------------------------
// Longformer forward for gfx1250 (MI455X).  All matmuls use
// v_wmma_f32_16x16x32_bf16 (wave32, 16x16 C tile per wave).  The main GEMM
// stages tiles into LDS with GLOBAL_LOAD_ASYNC_TO_LDS_B128 (ASYNCcnt),
// double-buffered and overlapped with WMMA compute.
//
#define kB   2
#define kS   4096
#define kDM  512
#define kNH  8
#define kDH  64
#define kNL  4
#define kDFF 1024
#define kW   256           // sliding window half-width (DM/2)
#define kBS  (kB * kS)     // 8192 token rows
#define kEC  1472          // sum(EMB)
#define kNEG (-1.0e9f)

typedef __attribute__((ext_vector_type(16))) __bf16 v16bf;
typedef __attribute__((ext_vector_type(8)))  float  v8f;

// --- WMMA helpers -----------------------------------------------------------
__device__ inline v8f wmma_bf16(v16bf a, v16bf b, v8f c) {
  return __builtin_amdgcn_wmma_f32_16x16x32_bf16(
      /*neg_a=*/false, a, /*neg_b=*/false, b,
      /*c_mod=*/(short)0, c, /*reuse_a=*/false, /*reuse_b=*/false);
}

// Fragment load: 16 bf16 elements as two contiguous 16-byte chunks at
// p[0..7] and p[16..23].  (K pattern: k = hi*8 + i%8 + (i/8)*16.)
__device__ inline v16bf load_frag(const __bf16* p) {
  union { v16bf v; uint4 q[2]; } u;
  u.q[0] = *(const uint4*)(p);
  u.q[1] = *(const uint4*)(p + 16);
  return u.v;
}

__device__ inline v16bf zero_frag() {
  union { v16bf v; uint4 q[2]; } u;
  u.q[0] = make_uint4(0u, 0u, 0u, 0u);
  u.q[1] = make_uint4(0u, 0u, 0u, 0u);
  return u.v;
}

// Async copy of 16 bytes global -> LDS (per lane).  lds_off is the workgroup
// LDS byte offset (= low 32 bits of the generic pointer for shared memory).
__device__ inline void async_copy16(unsigned lds_off, const __bf16* g) {
  asm volatile("global_load_async_to_lds_b128 %0, %1, off"
               :: "v"(lds_off), "v"(g) : "memory");
}
__device__ inline unsigned lds_off32(const void* p) {
  return (unsigned)(size_t)p;     // generic = {shared_aperture, lds_off[31:0]}
}

// ---------------------------------------------------------------------------
// Tiled GEMM: Cf/Cb[M,N] = act( A[M,K](bf16) * WT[N,K](bf16)^T + bias )
// Block = 256 thr = 8 waves.  Block tile 128(M) x 64(N); each wave computes
// 16x64 (4 accumulators).  K chunks of 64 staged to LDS via async-to-LDS,
// double buffered.  Requires M%128==0, N%64==0, K%64==0.
// act: 0 = none, 1 = exact GELU.
__global__ __launch_bounds__(256)
void gemm_bf16_kernel(const __bf16* __restrict__ A, const __bf16* __restrict__ WT,
                      const float* __restrict__ bias,
                      float* __restrict__ Cf, __bf16* __restrict__ Cb,
                      int M, int N, int K, int act) {
  __shared__ __align__(16) __bf16 sA[2][128 * 64];   // 2 x 16 KB
  __shared__ __align__(16) __bf16 sB[2][64 * 64];    // 2 x  8 KB
  const int tid  = threadIdx.x;
  const int lane = tid & 31;
  const int wv   = tid >> 5;
  const int tm   = blockIdx.y * 128;
  const int tn   = blockIdx.x * 64;
  const int hi   = lane >> 4;
  const int l15  = lane & 15;

  // Stage one 64-wide K chunk of the A (128 rows) and B (64 cols) tiles.
  // 6 async b128 ops per thread per stage.
  auto stage = [&](int buf, int k0) {
    #pragma unroll
    for (int i = 0; i < 4; ++i) {                      // A: 1024 transfers
      int idx = i * 256 + tid;
      int r = idx >> 3, pt = idx & 7;
      const __bf16* g = A + (size_t)(tm + r) * K + k0 + pt * 8;
      async_copy16(lds_off32(&sA[buf][r * 64 + pt * 8]), g);
    }
    #pragma unroll
    for (int i = 0; i < 2; ++i) {                      // B: 512 transfers
      int idx = i * 256 + tid;
      int cc = idx >> 3, pt = idx & 7;
      const __bf16* g = WT + (size_t)(tn + cc) * K + k0 + pt * 8;
      async_copy16(lds_off32(&sB[buf][cc * 64 + pt * 8]), g);
    }
  };

  v8f acc[4] = {{}, {}, {}, {}};
  const int nkc = K >> 6;
  stage(0, 0);
  for (int kc = 0; kc < nkc; ++kc) {
    const int cur = kc & 1;
    if (kc + 1 < nkc) {
      stage(cur ^ 1, (kc + 1) * 64);
      asm volatile("s_wait_asynccnt 6" ::: "memory");  // chunk kc complete
    } else {
      asm volatile("s_wait_asynccnt 0" ::: "memory");
    }
    __syncthreads();
    const __bf16* Abase = &sA[cur][(wv * 16 + l15) * 64];
    const __bf16* Bbase = &sB[cur][l15 * 64];
    #pragma unroll
    for (int kk = 0; kk < 64; kk += 32) {
      v16bf a = load_frag(Abase + kk + hi * 8);
      #pragma unroll
      for (int t = 0; t < 4; ++t) {
        v16bf b = load_frag(Bbase + t * 16 * 64 + kk + hi * 8);
        acc[t] = wmma_bf16(a, b, acc[t]);
      }
    }
    __syncthreads();   // protect buffer reuse (overwritten by next iter's stage)
  }

  #pragma unroll
  for (int t = 0; t < 4; ++t) {
    #pragma unroll
    for (int r = 0; r < 8; ++r) {
      int mm = tm + wv * 16 + hi * 8 + r;   // C layout: M = r + hi*8
      int nn = tn + t * 16 + l15;           //           N = lane & 15
      float v = acc[t][r];
      if (bias) v += bias[nn];
      if (act == 1) v = 0.5f * v * (1.0f + erff(v * 0.70710678118f));
      if (Cf) Cf[(size_t)mm * N + nn] = v;
      if (Cb) Cb[(size_t)mm * N + nn] = (__bf16)v;
    }
  }
}

// ---------------------------------------------------------------------------
// Weight convert+transpose: WT[n*K + k] = (bf16) W[k*N + n]
__global__ void convert_wT_kernel(const float* __restrict__ w, __bf16* __restrict__ wt,
                                  int K, int N) {
  int idx = blockIdx.x * blockDim.x + threadIdx.x;
  if (idx >= N * K) return;
  int n = idx / K, k = idx - n * K;
  wt[idx] = (__bf16)w[(size_t)k * N + n];
}

// ---------------------------------------------------------------------------
// Embedding gather + concat (scaled by sqrt(EMB_f)), bf16 output [kBS x kEC].
__global__ __launch_bounds__(256)
void embed_concat_kernel(const int* __restrict__ data,
                         const float* e0, const float* e1, const float* e2,
                         const float* e3, const float* e4, const float* e5,
                         __bf16* __restrict__ ea) {
  const int t = blockIdx.x;                       // token row 0..kBS-1
  const int   off[6] = {0, 128, 384, 448, 960, 1216};
  const int   esz[6] = {128, 256, 64, 512, 256, 256};
  const float* ep[6] = {e0, e1, e2, e3, e4, e5};
  int idxs[6];
  #pragma unroll
  for (int f = 0; f < 6; ++f) idxs[f] = data[(size_t)t * 6 + f];
  for (int j = threadIdx.x; j < kEC; j += blockDim.x) {
    int f = (j < 128) ? 0 : (j < 384) ? 1 : (j < 448) ? 2
          : (j < 960) ? 3 : (j < 1216) ? 4 : 5;
    int jj = j - off[f];
    float v = ep[f][(size_t)idxs[f] * esz[f] + jj] * sqrtf((float)esz[f]);
    ea[(size_t)t * kEC + j] = (__bf16)v;
  }
}

// ---------------------------------------------------------------------------
// LayerNorm over DM=512:  in = a [+ res] [+ pos_emb[s+2] + tok];
// out f32 x and bf16 xh.  One block (256 thr) per row.
__global__ __launch_bounds__(256)
void ln_kernel(const float* __restrict__ a, const float* __restrict__ res,
               const float* __restrict__ pos, const float* __restrict__ tok,
               const float* __restrict__ gs, const float* __restrict__ gb,
               float* __restrict__ xo, __bf16* __restrict__ xh) {
  const int row = blockIdx.x;          // 0..kBS-1
  const int s   = row & (kS - 1);
  __shared__ float red[256];
  float v[2];
  #pragma unroll
  for (int i = 0; i < 2; ++i) {
    int d = threadIdx.x + i * 256;
    float t = a[(size_t)row * kDM + d];
    if (res) t += res[(size_t)row * kDM + d];
    if (pos) t += pos[(size_t)(s + 2) * kDM + d] + tok[d];
    v[i] = t;
  }
  red[threadIdx.x] = v[0] + v[1];
  __syncthreads();
  for (int st = 128; st > 0; st >>= 1) {
    if (threadIdx.x < st) red[threadIdx.x] += red[threadIdx.x + st];
    __syncthreads();
  }
  float mean = red[0] * (1.0f / kDM);
  __syncthreads();
  float d0 = v[0] - mean, d1 = v[1] - mean;
  red[threadIdx.x] = d0 * d0 + d1 * d1;
  __syncthreads();
  for (int st = 128; st > 0; st >>= 1) {
    if (threadIdx.x < st) red[threadIdx.x] += red[threadIdx.x + st];
    __syncthreads();
  }
  float rstd = rsqrtf(red[0] * (1.0f / kDM) + 1e-12f);
  #pragma unroll
  for (int i = 0; i < 2; ++i) {
    int d = threadIdx.x + i * 256;
    float o = (v[i] - mean) * rstd * gs[d] + gb[d];
    xo[(size_t)row * kDM + d] = o;
    xh[(size_t)row * kDM + d] = (__bf16)o;
  }
}

// ---------------------------------------------------------------------------
// Reshape [kBS x DM] f32 -> bf16 head layouts.
// mode 0: q[b,h,s,d] *= 1/sqrt(DH);  mode 1: k[b,h,s,d];  mode 2: vT[b,h,d,s]
__global__ __launch_bounds__(256)
void reshape_kernel(const float* __restrict__ t, __bf16* __restrict__ dst, int mode) {
  size_t i = (size_t)blockIdx.x * 256 + threadIdx.x;   // over kBS*kDM
  if (i >= (size_t)kBS * kDM) return;
  int dd  = (int)(i & (kDM - 1));
  size_t row = i >> 9;                                  // b*kS + s
  int b = (int)(row >> 12), s = (int)(row & (kS - 1));
  int h = dd >> 6, d = dd & 63;
  float v = t[i];
  if (mode == 0)
    dst[(((size_t)b * kNH + h) * kS + s) * kDH + d] = (__bf16)(v * 0.125f);
  else if (mode == 1)
    dst[(((size_t)b * kNH + h) * kS + s) * kDH + d] = (__bf16)v;
  else
    dst[(((size_t)b * kNH + h) * kDH + d) * kS + s] = (__bf16)v;
}

// ---------------------------------------------------------------------------
// Sliding-window attention. One wave (32 thr) handles 16 query rows of one
// chunk: scores 16x768 via WMMA (QK^T), masked softmax in LDS, PV via WMMA.
// grid.x = B*NH*NC*(W/16) = 4096.
__global__ __launch_bounds__(32)
void attn_kernel(const __bf16* __restrict__ q, const __bf16* __restrict__ k,
                 const __bf16* __restrict__ vt, const float* __restrict__ amask,
                 __bf16* __restrict__ o) {
  __shared__ __align__(16) float sS[16 * 768];   // 48 KB
  const int lane = threadIdx.x;
  const int qt = blockIdx.x & 15;
  const int c  = (blockIdx.x >> 4) & 15;
  const int h  = (blockIdx.x >> 8) & 7;
  const int b  = blockIdx.x >> 11;
  const int hi = lane >> 4, l15 = lane & 15;
  const size_t bh = (size_t)b * kNH + h;

  const size_t qbase = (bh * kS + (size_t)c * kW + qt * 16 + l15) * kDH;
  v16bf qa0 = load_frag(q + qbase + hi * 8);
  v16bf qa1 = load_frag(q + qbase + 32 + hi * 8);

  // ---- scores: 48 key tiles of 16 -----------------------------------------
  for (int nt = 0; nt < 48; ++nt) {
    int col = nt * 16 + l15;                  // key column within 3W window
    long kp = (long)c * kW - kW + col;        // absolute key position
    v16bf kb0, kb1;
    if (kp >= 0 && kp < kS) {
      const size_t kbase = (bh * kS + (size_t)kp) * kDH;
      kb0 = load_frag(k + kbase + hi * 8);
      kb1 = load_frag(k + kbase + 32 + hi * 8);
    } else {
      kb0 = zero_frag(); kb1 = zero_frag();
    }
    v8f acc = {};
    acc = wmma_bf16(qa0, kb0, acc);
    acc = wmma_bf16(qa1, kb1, acc);
    #pragma unroll
    for (int r = 0; r < 8; ++r) {
      int m  = hi * 8 + r;                    // query row within tile
      int qi = qt * 16 + m;                   // query row within chunk
      int kj = nt * 16 + l15;
      long kpp = (long)c * kW - kW + kj;
      bool ok = (kj >= qi) && (kj <= qi + 2 * kW) &&
                (kpp >= 0) && (kpp < kS) &&
                (amask[(size_t)b * kS + kpp] > 0.0f);
      sS[m * 768 + kj] = ok ? acc[r] : kNEG;
    }
  }
  __syncthreads();

  // ---- softmax: lane pair (l, l^16) shares row l&15 ------------------------
  {
    const int m = l15;
    float mx = -3.0e38f;
    for (int j = hi * 384; j < hi * 384 + 384; ++j)
      mx = fmaxf(mx, sS[m * 768 + j]);
    mx = fmaxf(mx, __shfl_xor(mx, 16, 32));
    float sum = 0.0f;
    for (int j = hi * 384; j < hi * 384 + 384; ++j) {
      float e = __expf(sS[m * 768 + j] - mx);
      sS[m * 768 + j] = e;
      sum += e;
    }
    sum += __shfl_xor(sum, 16, 32);
    float inv = 1.0f / sum;
    for (int j = hi * 384; j < hi * 384 + 384; ++j)
      sS[m * 768 + j] *= inv;
  }
  __syncthreads();

  // ---- PV: out[16 x 64] = P[16 x 768] * V[768 x 64] ------------------------
  for (int dt = 0; dt < 4; ++dt) {
    v8f acc = {};
    const int d = dt * 16 + l15;
    const __bf16* vp = vt + (bh * kDH + d) * kS;
    for (int ks = 0; ks < 24; ++ks) {
      const float* pr = sS + l15 * 768 + ks * 32 + hi * 8;
      union { v16bf v; __bf16 e[16]; } ua;
      #pragma unroll
      for (int i = 0; i < 8; ++i) {
        ua.e[i]     = (__bf16)pr[i];
        ua.e[8 + i] = (__bf16)pr[16 + i];
      }
      long kbase = (long)c * kW - kW + ks * 32 + hi * 8;
      union { v16bf v; uint4 qq[2]; __bf16 e[16]; } ub;
      if (kbase >= 0 && kbase + 7 < kS) {
        ub.qq[0] = *(const uint4*)(vp + kbase);
      } else {
        #pragma unroll
        for (int i = 0; i < 8; ++i) {
          long kp = kbase + i;
          ub.e[i] = (kp >= 0 && kp < kS) ? vp[kp] : (__bf16)0.0f;
        }
      }
      if (kbase + 16 >= 0 && kbase + 23 < kS) {
        ub.qq[1] = *(const uint4*)(vp + kbase + 16);
      } else {
        #pragma unroll
        for (int i = 0; i < 8; ++i) {
          long kp = kbase + 16 + i;
          ub.e[8 + i] = (kp >= 0 && kp < kS) ? vp[kp] : (__bf16)0.0f;
        }
      }
      acc = wmma_bf16(ua.v, ub.v, acc);
    }
    #pragma unroll
    for (int r = 0; r < 8; ++r) {
      int s = c * kW + qt * 16 + hi * 8 + r;
      o[((size_t)b * kS + s) * kDM + h * kDH + d] = (__bf16)acc[r];
    }
  }
}

// ---------------------------------------------------------------------------
// Fold head i:  weff[d] = sum_j pw[d,j]*ew[j];  ceff = sum_j pb[j]*ew[j] + eb
__global__ __launch_bounds__(256)
void head_eff_kernel(const float* __restrict__ pw, const float* __restrict__ pb,
                     const float* __restrict__ ew, const float* __restrict__ eb,
                     int n, float* __restrict__ weff, float* __restrict__ ceff) {
  int d = blockIdx.x * blockDim.x + threadIdx.x;
  if (d < kDM) {
    float s = 0.0f;
    for (int j = 0; j < n; ++j) s += pw[(size_t)d * n + j] * ew[j];
    weff[d] = s;
  }
  if (blockIdx.x == 0 && threadIdx.x == 0) {
    float s = 0.0f;
    for (int j = 0; j < n; ++j) s += pb[j] * ew[j];
    *ceff = s + eb[0];
  }
}

// score[b] += sigmoid(mean_s(x·weff) + ceff) / 6.  grid = B*6.
__global__ __launch_bounds__(256)
void head_score_kernel(const float* __restrict__ x, const float* __restrict__ weff,
                       const float* __restrict__ ceff, float* __restrict__ out) {
  const int b = blockIdx.x / 6, hh = blockIdx.x % 6;
  const float* wf = weff + hh * kDM;
  __shared__ float red[256];
  float acc = 0.0f;
  for (int s = threadIdx.x; s < kS; s += 256) {
    const float* xr = x + ((size_t)b * kS + s) * kDM;
    float dot = 0.0f;
    for (int d = 0; d < kDM; ++d) dot += xr[d] * wf[d];
    acc += dot;
  }
  red[threadIdx.x] = acc;
  __syncthreads();
  for (int st = 128; st > 0; st >>= 1) {
    if (threadIdx.x < st) red[threadIdx.x] += red[threadIdx.x + st];
    __syncthreads();
  }
  if (threadIdx.x == 0) {
    float hid = red[0] * (1.0f / kS) + ceff[hh];
    float sig = 1.0f / (1.0f + __expf(-hid));
    atomicAdd(&out[b], sig * (1.0f / 6.0f));
  }
}

__global__ void zero_kernel(float* p, int n) {
  int i = blockIdx.x * blockDim.x + threadIdx.x;
  if (i < n) p[i] = 0.0f;
}

// ---------------------------------------------------------------------------
extern "C" void kernel_launch(void* const* d_in, const int* in_sizes, int n_in,
                              void* d_out, int out_size, void* d_ws, size_t ws_size,
                              hipStream_t stream) {
  (void)in_sizes; (void)n_in; (void)ws_size;
  const int*   data   = (const int*)d_in[0];
  const float* lmask  = (const float*)d_in[1];
  const float* emb[6];
  for (int i = 0; i < 6; ++i) emb[i] = (const float*)d_in[2 + i];
  const float* proj_w = (const float*)d_in[8];
  const float* proj_b = (const float*)d_in[9];
  const float* pos    = (const float*)d_in[10];
  const float* tok    = (const float*)d_in[11];
  const float* elns   = (const float*)d_in[12];
  const float* elnb   = (const float*)d_in[13];
  const int LBASE = 14, HBASE = 14 + kNL * 16;
  const int NTOK[6] = {56, 135, 18, 87, 18, 25};

  // --- workspace layout (256B aligned), ~152 MB total ----------------------
  char* wp = (char*)d_ws;
  auto alloc = [&](size_t bytes) -> char* {
    char* p = wp; wp += (bytes + 255) & ~(size_t)255; return p;
  };
  __bf16* ea      = (__bf16*)alloc((size_t)kBS * kEC * 2);
  __bf16* wT_proj = (__bf16*)alloc((size_t)kDM * kEC * 2);   // [512][1472]
  __bf16* wTq[kNL], *wTk[kNL], *wTv[kNL], *wTo[kNL], *wTi[kNL], *wTo2[kNL];
  for (int l = 0; l < kNL; ++l) {
    wTq[l]  = (__bf16*)alloc((size_t)kDM * kDM * 2);
    wTk[l]  = (__bf16*)alloc((size_t)kDM * kDM * 2);
    wTv[l]  = (__bf16*)alloc((size_t)kDM * kDM * 2);
    wTo[l]  = (__bf16*)alloc((size_t)kDM * kDM * 2);
    wTi[l]  = (__bf16*)alloc((size_t)kDFF * kDM * 2);        // [1024][512]
    wTo2[l] = (__bf16*)alloc((size_t)kDM * kDFF * 2);        // [512][1024]
  }
  float*  x    = (float*) alloc((size_t)kBS * kDM * 4);
  __bf16* xh   = (__bf16*)alloc((size_t)kBS * kDM * 2);
  float*  tmp  = (float*) alloc((size_t)kBS * kDFF * 4);
  __bf16* hidb = (__bf16*)alloc((size_t)kBS * kDFF * 2);
  __bf16* qb   = (__bf16*)alloc((size_t)kBS * kDM * 2);
  __bf16* kb   = (__bf16*)alloc((size_t)kBS * kDM * 2);
  __bf16* vtb  = (__bf16*)alloc((size_t)kBS * kDM * 2);
  __bf16* ob   = (__bf16*)alloc((size_t)kBS * kDM * 2);
  float*  weff = (float*) alloc(6 * kDM * 4);
  float*  ceff = (float*) alloc(6 * 4);

  zero_kernel<<<1, 32, 0, stream>>>((float*)d_out, out_size);

  // --- convert + transpose all weights to bf16 -----------------------------
  auto conv = [&](const float* w, __bf16* wt, int K, int N) {
    int n = N * K;
    convert_wT_kernel<<<(n + 255) / 256, 256, 0, stream>>>(w, wt, K, N);
  };
  conv(proj_w, wT_proj, kEC, kDM);
  for (int l = 0; l < kNL; ++l) {
    const int base = LBASE + l * 16;
    conv((const float*)d_in[base + 0],  wTq[l],  kDM, kDM);
    conv((const float*)d_in[base + 2],  wTk[l],  kDM, kDM);
    conv((const float*)d_in[base + 4],  wTv[l],  kDM, kDM);
    conv((const float*)d_in[base + 6],  wTo[l],  kDM, kDM);
    conv((const float*)d_in[base + 10], wTi[l],  kDM, kDFF);
    conv((const float*)d_in[base + 12], wTo2[l], kDFF, kDM);
  }

  auto gemm = [&](const __bf16* A, const __bf16* WT, const float* bias,
                  float* Cf, __bf16* Cb, int M, int N, int K, int act) {
    dim3 grid(N / 64, M / 128);
    gemm_bf16_kernel<<<grid, 256, 0, stream>>>(A, WT, bias, Cf, Cb, M, N, K, act);
  };

  // --- embeddings -> projection -> LN --------------------------------------
  embed_concat_kernel<<<kBS, 256, 0, stream>>>(data, emb[0], emb[1], emb[2],
                                               emb[3], emb[4], emb[5], ea);
  gemm(ea, wT_proj, proj_b, tmp, nullptr, kBS, kDM, kEC, 0);
  ln_kernel<<<kBS, 256, 0, stream>>>(tmp, nullptr, pos, tok, elns, elnb, x, xh);

  // --- transformer layers ---------------------------------------------------
  for (int l = 0; l < kNL; ++l) {
    const int base = LBASE + l * 16;
    const float* bq   = (const float*)d_in[base + 1];
    const float* bk   = (const float*)d_in[base + 3];
    const float* bv   = (const float*)d_in[base + 5];
    const float* bo   = (const float*)d_in[base + 7];
    const float* ln1s = (const float*)d_in[base + 8];
    const float* ln1b = (const float*)d_in[base + 9];
    const float* bi   = (const float*)d_in[base + 11];
    const float* bo2  = (const float*)d_in[base + 13];
    const float* ln2s = (const float*)d_in[base + 14];
    const float* ln2b = (const float*)d_in[base + 15];
    const int nblk = (int)(((size_t)kBS * kDM + 255) / 256);

    gemm(xh, wTq[l], bq, tmp, nullptr, kBS, kDM, kDM, 0);
    reshape_kernel<<<nblk, 256, 0, stream>>>(tmp, qb, 0);
    gemm(xh, wTk[l], bk, tmp, nullptr, kBS, kDM, kDM, 0);
    reshape_kernel<<<nblk, 256, 0, stream>>>(tmp, kb, 1);
    gemm(xh, wTv[l], bv, tmp, nullptr, kBS, kDM, kDM, 0);
    reshape_kernel<<<nblk, 256, 0, stream>>>(tmp, vtb, 2);

    attn_kernel<<<kB * kNH * (kS / kW) * (kW / 16), 32, 0, stream>>>(
        qb, kb, vtb, lmask, ob);

    gemm(ob, wTo[l], bo, tmp, nullptr, kBS, kDM, kDM, 0);
    ln_kernel<<<kBS, 256, 0, stream>>>(tmp, x, nullptr, nullptr, ln1s, ln1b, x, xh);

    gemm(xh, wTi[l], bi, nullptr, hidb, kBS, kDFF, kDM, 1);     // GELU
    gemm(hidb, wTo2[l], bo2, tmp, nullptr, kBS, kDM, kDFF, 0);
    ln_kernel<<<kBS, 256, 0, stream>>>(tmp, x, nullptr, nullptr, ln2s, ln2b, x, xh);
  }

  // --- folded heads ---------------------------------------------------------
  for (int i = 0; i < 6; ++i) {
    const int hb = HBASE + i * 4;
    head_eff_kernel<<<2, 256, 0, stream>>>(
        (const float*)d_in[hb + 0], (const float*)d_in[hb + 1],
        (const float*)d_in[hb + 2], (const float*)d_in[hb + 3],
        NTOK[i], weff + i * kDM, ceff + i);
  }
  head_score_kernel<<<kB * 6, 256, 0, stream>>>(x, weff, ceff, (float*)d_out);
}
